// LIIF_rendercopy_88493506167388
// MI455X (gfx1250) — matrix operand order repaired and checked by reading, not compiled, not run on hardware
//
#include <hip/hip_runtime.h>
#include <hip/hip_bf16.h>

typedef __attribute__((ext_vector_type(16))) _Float16 v16h;
typedef __attribute__((ext_vector_type(8)))  _Float16 v8h;
typedef __attribute__((ext_vector_type(2)))  _Float16 v2h;
typedef __attribute__((ext_vector_type(8)))  float    v8f;

// Problem sizes (fixed by setup_inputs: N=4, C=64, Hf=Wf=64, h=w=256, cell_scale=1)
#define NB        4
#define CFEAT     64
#define HF        64
#define HOUT      256
#define QTOT      65536      // HOUT*HOUT
#define ROWS_TOT  262144     // NB*QTOT
#define TILE_ROWS 128        // 8 waves * 16 rows

// f16 weight-fragment offsets (in halves) inside d_ws
#define L0_OFF 0             // 3 ktiles * 16 ntiles * 512
#define L1_OFF 24576
#define L2_OFF 90112
#define L3_OFF 155648
#define L4_OFF 221184        // 8 ktiles * 1 ntile * 512
#define WTOT   225280

// ---------------------------------------------------------------------------
// Pre-pass: convert fp32 weights to f16 in WMMA B-fragment order.
// B is 32(K)x16(N), packed in contiguous k order (this code's choice; load_a uses the same):
//   lane l, half e (0..15 across 8 VGPRs): K = 32*ktile + 16*(l>=16) + e,
//   N = 16*ntile + (l & 15).
// ---------------------------------------------------------------------------
__global__ void prep_weights(const float* __restrict__ w0,
                             const float* __restrict__ w1,
                             const float* __restrict__ w2,
                             const float* __restrict__ w3,
                             const float* __restrict__ w4,
                             _Float16* __restrict__ outw) {
    int tid = blockIdx.x * 256 + threadIdx.x;
    if (tid >= WTOT) return;
    const float* w; int rem, Kreal, Nreal, nt;
    if (tid < L1_OFF)        { w = w0; rem = tid;          Kreal = 68;  Nreal = 256; nt = 16; }
    else if (tid < L2_OFF)   { w = w1; rem = tid - L1_OFF; Kreal = 256; Nreal = 256; nt = 16; }
    else if (tid < L3_OFF)   { w = w2; rem = tid - L2_OFF; Kreal = 256; Nreal = 256; nt = 16; }
    else if (tid < L4_OFF)   { w = w3; rem = tid - L3_OFF; Kreal = 256; Nreal = 256; nt = 16; }
    else                     { w = w4; rem = tid - L4_OFF; Kreal = 256; Nreal = 3;   nt = 1;  }
    int frag  = rem >> 9;          // /512 halves per fragment
    int el    = rem & 511;
    int lane  = el >> 4;
    int e     = el & 15;
    int ktile = frag / nt;
    int ntile = frag - ktile * nt;
    int K = ktile * 32 + ((lane >= 16) ? 16 : 0) + e;
    int N = ntile * 16 + (lane & 15);
    float v = 0.0f;
    if (K < Kreal && N < Nreal) v = w[K * Nreal + N];
    outw[tid] = (_Float16)v;
}

// ---------------------------------------------------------------------------
// Geometry helpers (h=w=256, Hf=Wf=64, cell_scale=1)
// ---------------------------------------------------------------------------
__device__ __forceinline__ float clampf(float x, float lo, float hi) {
    return fminf(fmaxf(x, lo), hi);
}

__device__ __forceinline__ void geom(int q, int vx, int vy,
                                     int& iy, int& ix, float& rel0, float& rel1) {
    int y = q >> 8, x = q & 255;
    float yf = (2.0f * (float)y + 1.0f) * (1.0f / 256.0f) - 1.0f;
    float xf = (2.0f * (float)x + 1.0f) * (1.0f / 256.0f) - 1.0f;
    float c0 = clampf(yf + (float)vx * (1.0f / 64.0f) + 1e-6f, -1.0f + 1e-6f, 1.0f - 1e-6f);
    float c1 = clampf(xf + (float)vy * (1.0f / 64.0f) + 1e-6f, -1.0f + 1e-6f, 1.0f - 1e-6f);
    float fiy = rintf(((c0 + 1.0f) * 64.0f - 1.0f) * 0.5f);
    float fix = rintf(((c1 + 1.0f) * 64.0f - 1.0f) * 0.5f);
    iy = (int)clampf(fiy, 0.0f, 63.0f);
    ix = (int)clampf(fix, 0.0f, 63.0f);
    float qc0 = (2.0f * (float)iy + 1.0f) * (1.0f / 64.0f) - 1.0f;
    float qc1 = (2.0f * (float)ix + 1.0f) * (1.0f / 64.0f) - 1.0f;
    rel0 = (yf - qc0) * 64.0f;
    rel1 = (xf - qc1) * 64.0f;
}

__device__ __forceinline__ float area_of(int q, int ss) {
    int vx = (ss & 2) ? 1 : -1;
    int vy = (ss & 1) ? 1 : -1;
    int iy, ix; float r0, r1;
    geom(q, vx, vy, iy, ix, r0, r1);
    return fabsf(r0 * r1) + 1e-9f;
}

// A-fragment (16x32 f16, M = lane&15, K halves selected by lane>=16):
__device__ __forceinline__ v16h load_a(const _Float16* arow, int kbase, int hi) {
    v16h a;
    *(v8h*)&a         = *(const v8h*)(arow + kbase + hi * 16);
    *(((v8h*)&a) + 1) = *(const v8h*)(arow + kbase + hi * 16 + 8);
    return a;
}

__device__ __forceinline__ v2h pack_relu(float e0, float e1) {
    return __builtin_bit_cast(v2h, __builtin_amdgcn_cvt_pkrtz(fmaxf(e0, 0.0f),
                                                              fmaxf(e1, 0.0f)));
}

// ---------------------------------------------------------------------------
// Fused LIIF kernel: 8 waves/block, each wave owns 16 query rows end-to-end.
// ---------------------------------------------------------------------------
__global__ __launch_bounds__(256) void liif_fused(
    const float* __restrict__ feat,
    const float* __restrict__ b0, const float* __restrict__ b1,
    const float* __restrict__ b2, const float* __restrict__ b3,
    const float* __restrict__ b4,
    const _Float16* __restrict__ wh,
    float* __restrict__ out)
{
    __shared__ _Float16 stage[8][16][256];   // 64 KB: per-wave 16x256 activation slab

    const int lane = threadIdx.x & 31;
    const int wid  = threadIdx.x >> 5;
    const int ln   = lane & 15;
    const int hi   = lane >> 4;
    const int rowbase = blockIdx.x * TILE_ROWS + wid * 16;
    const int n = rowbase >> 16;             // Q = 65536, slabs never straddle batches
    const float* featn = feat + n * (CFEAT * HF * HF);
    const _Float16* arow = &stage[wid][ln][0];

    const int    loff[4] = {L0_OFF, L1_OFF, L2_OFF, L3_OFF};
    const int    ktls[4] = {3, 8, 8, 8};
    const float* bls[4]  = {b0, b1, b2, b3};

    float outacc[8];
#pragma unroll
    for (int v = 0; v < 8; ++v) outacc[v] = 0.0f;

    for (int s = 0; s < 4; ++s) {
        const int vx = (s & 2) ? 1 : -1;
        const int vy = (s & 1) ? 1 : -1;

        // ---- build layer-0 input (96 cols: 64 feat + rel(2) + cell(2) + pad) ----
        {
            int row = ln;                        // this lane writes row (lane&15)
            int q = (rowbase + row) & (QTOT - 1);
            int iy, ix; float r0, r1;
            geom(q, vx, vy, iy, ix, r0, r1);
            int fbase = iy * 64 + ix;
            int cstart = hi * 48;
            for (int t = 0; t < 48; ++t) {
                int c = cstart + t;
                float val;
                if (c < 64)       val = featn[c * 4096 + fbase];
                else if (c == 64) val = r0;
                else if (c == 65) val = r1;
                else if (c < 68)  val = 0.5f;    // rel_cell = (2/256)*64
                else              val = 0.0f;
                stage[wid][row][c] = (_Float16)val;
            }
        }
        __syncthreads();

        // ---- hidden layers 0..3 (rolled; per-layer params from lookup) ----
        for (int l = 0; l < 4; ++l) {
            const _Float16* wl  = wh + loff[l];
            const float*    bl  = bls[l];
            const int       ktl = ktls[l];

            v2h packo[64];   // 16 ntiles x 4 packed row-pairs, built in 2 halves
#pragma unroll
            for (int half = 0; half < 2; ++half) {
                v8f acc[8];
#pragma unroll
                for (int t = 0; t < 8; ++t) {
                    float bsc = bl[(half * 8 + t) * 16 + ln];
#pragma unroll
                    for (int v = 0; v < 8; ++v) acc[t][v] = bsc;
                }
                for (int kt = 0; kt < ktl; ++kt) {          // rolled K loop
                    v16h a = load_a(arow, kt * 32, hi);
#pragma unroll
                    for (int t = 0; t < 8; ++t) {
                        int nt = half * 8 + t;
                        v16h b = *(const v16h*)(wl + ((kt * 16 + nt) * 32 + lane) * 16);
                        acc[t] = __builtin_amdgcn_wmma_f32_16x16x32_f16(
                            false, a, false, b, (short)0, acc[t], false, false);
                    }
                }
#pragma unroll
                for (int t = 0; t < 8; ++t)
#pragma unroll
                    for (int j = 0; j < 4; ++j)
                        packo[(half * 8 + t) * 4 + j] =
                            pack_relu(acc[t][2 * j], acc[t][2 * j + 1]);
            }
            __syncthreads();   // all A-frag reads done before overwriting slab
#pragma unroll
            for (int nt = 0; nt < 16; ++nt) {
                int col = nt * 16 + ln;
#pragma unroll
                for (int j = 0; j < 4; ++j) {
                    stage[wid][hi * 8 + 2 * j][col]     = packo[nt * 4 + j].x;
                    stage[wid][hi * 8 + 2 * j + 1][col] = packo[nt * 4 + j].y;
                }
            }
            __syncthreads();
        }

        // ---- final layer 256 -> 3 (padded to 16 cols) ----
        {
            float bsc = (ln < 3) ? b4[ln] : 0.0f;
            v8f acc;
#pragma unroll
            for (int v = 0; v < 8; ++v) acc[v] = bsc;
            for (int kt = 0; kt < 8; ++kt) {                // rolled K loop
                v16h a = load_a(arow, kt * 32, hi);
                v16h b = *(const v16h*)(wh + L4_OFF + (kt * 32 + lane) * 16);
                acc = __builtin_amdgcn_wmma_f32_16x16x32_f16(
                    false, a, false, b, (short)0, acc, false, false);
            }
            // ---- blend: pred_s weighted by area[3-s] / tot ----
#pragma unroll
            for (int v = 0; v < 8; ++v) {
                int r16 = hi * 8 + v;                 // D layout: M = v + 8*(lane>=16)
                int q = (rowbase + r16) & (QTOT - 1);
                float a0 = area_of(q, 0), a1 = area_of(q, 1);
                float a2 = area_of(q, 2), a3 = area_of(q, 3);
                float tot = a0 + a1 + a2 + a3;
                float ar = (s == 0) ? a3 : (s == 1) ? a2 : (s == 2) ? a1 : a0;
                outacc[v] += acc[v] * (ar / tot);
            }
        }
        __syncthreads();
    }

    // ---- write output (N,3,h,w); lanes with ln<3 own channel ln ----
    if (ln < 3) {
#pragma unroll
        for (int v = 0; v < 8; ++v) {
            int row = rowbase + hi * 8 + v;
            int q = row & (QTOT - 1);
            int y = q >> 8, x = q & 255;
            out[(((n * 3 + ln) * HOUT) + y) * HOUT + x] = outacc[v];
        }
    }
}

// ---------------------------------------------------------------------------
extern "C" void kernel_launch(void* const* d_in, const int* in_sizes, int n_in,
                              void* d_out, int out_size, void* d_ws, size_t ws_size,
                              hipStream_t stream) {
    (void)in_sizes; (void)n_in; (void)out_size; (void)ws_size;
    const float* feat = (const float*)d_in[0];
    const float* w0 = (const float*)d_in[1];  const float* b0 = (const float*)d_in[2];
    const float* w1 = (const float*)d_in[3];  const float* b1 = (const float*)d_in[4];
    const float* w2 = (const float*)d_in[5];  const float* b2 = (const float*)d_in[6];
    const float* w3 = (const float*)d_in[7];  const float* b3 = (const float*)d_in[8];
    const float* w4 = (const float*)d_in[9];  const float* b4 = (const float*)d_in[10];
    _Float16* wh = (_Float16*)d_ws;
    float* outp = (float*)d_out;

    prep_weights<<<(WTOT + 255) / 256, 256, 0, stream>>>(w0, w1, w2, w3, w4, wh);
    liif_fused<<<ROWS_TOT / TILE_ROWS, 256, 0, stream>>>(
        feat, b0, b1, b2, b3, b4, wh, outp);
}
